// Model_19456201851136
// MI455X (gfx1250) — matrix operand order repaired and checked by reading, not compiled, not run on hardware
//
#include <hip/hip_runtime.h>
#include <hip/hip_bf16.h>

typedef __attribute__((ext_vector_type(16))) _Float16 v16h;
typedef __attribute__((ext_vector_type(8)))  float    v8f;

#define GSIZE 200
#define G2    (GSIZE * GSIZE)
#define NPTS  128

constexpr float MIN_LON = 95.987f,  MIN_LAT = 31.3039f;
constexpr float MAX_LON = 109.4132f, MAX_LAT = 42.879f;
constexpr float DLON = (MAX_LON - MIN_LON) / (float)GSIZE;
constexpr float DLAT = (MAX_LAT - MIN_LAT) / (float)GSIZE;
constexpr float DEG2RAD = 0.017453292519943295f;
constexpr float EARTH_R = 6371.0f;

struct P {
  const float *x, *w1, *b1, *g1, *be1, *m1, *v1,
              *w2, *b2, *g2, *be2, *m2, *v2, *w3, *b3;
};

// Branch-free tanh: 1 - 2*rcp(exp(2x)+1).
// v_exp_f32 + v_rcp_f32, no EXEC divergence; saturates correctly at +-inf.
__device__ __forceinline__ float fast_tanh(float x) {
  const float e = __expf(2.0f * x);
  return fmaf(-2.0f, __builtin_amdgcn_rcpf(e + 1.0f), 1.0f);
}

// One block = one grid cell k, 128 threads = 128 points (4 wave32).
// Computes per-lane: predicted class, great-circle distance (km), time, real id.
__device__ __forceinline__ void model_block(int k, const P p,
                                            int& pid_o, float& gd_o,
                                            float& tim_o, int& rid_o)
{
  __shared__ _Float16 w2h[32][33];      // B matrix layer2 (K x N), f16
  __shared__ _Float16 w3h[32][17];      // B matrix layer3, N padded 5->16
  __shared__ float s1[32], t1[32], s2[32], t2[32], w1s[64], b3s[5];
  __shared__ _Float16 hbuf [NPTS][34];  // layer1 activations (rows x 32), padded
  __shared__ _Float16 hbuf2[NPTS][34];  // layer2 activations
  __shared__ float    lbuf [NPTS][6];   // logits (rows x 5), padded

  const int tid  = threadIdx.x;   // 0..127
  const int lane = tid & 31;
  const int wave = tid >> 5;

  // ---- stage weights: fold BN into scale/shift, convert GEMM weights to f16
  for (int c = tid; c < 1024; c += NPTS)
    w2h[c >> 5][c & 31] = (_Float16)p.w2[c];
  for (int c = tid; c < 512; c += NPTS) {
    const int r = c >> 4, col = c & 15;
    w3h[r][col] = (col < 5) ? (_Float16)p.w3[r * 5 + col] : (_Float16)0.0f;
  }
  if (tid < 32) {
    const float sa = p.g1[tid] * rsqrtf(p.v1[tid] + 1e-5f);
    s1[tid] = sa;
    t1[tid] = sa * (p.b1[tid] - p.m1[tid]) + p.be1[tid];
    const float sb = p.g2[tid] * rsqrtf(p.v2[tid] + 1e-5f);
    s2[tid] = sb;
    t2[tid] = sb * (p.b2[tid] - p.m2[tid]) + p.be2[tid];
    w1s[tid]      = p.w1[tid];
    w1s[32 + tid] = p.w1[32 + tid];
  }
  if (tid < 5) b3s[tid] = p.b3[tid];
  __syncthreads();

  // ---- haversine / great-circle distance (hw trans ops; args are < ~2 rad,
  //      so v_sin/v_cos need no range reduction)
  const int   gi   = k / GSIZE, gj = k - gi * GSIZE;
  const float lat1 = (MIN_LAT + DLAT * (float)gi) * DEG2RAD;
  const float lon1 = (MIN_LON + DLON * (float)gj) * DEG2RAD;
  const float lon2 = p.x[tid * 4 + 0] * DEG2RAD;
  const float lat2 = p.x[tid * 4 + 1] * DEG2RAD;
  const float tim  = p.x[tid * 4 + 2];
  const int   rid  = (int)p.x[tid * 4 + 3];

  const float dl   = lon2 - lon1;
  const float sdl  = __sinf(dl),   cdl  = __cosf(dl);
  const float sla1 = __sinf(lat1), cla1 = __cosf(lat1);
  const float sla2 = __sinf(lat2), cla2 = __cosf(lat2);
  const float ya = cla2 * sdl;
  const float yb = cla1 * sla2 - sla1 * cla2 * cdl;
  const float xx = sla1 * sla2 + cla1 * cla2 * cdl;
  const float gd = atan2f(sqrtf(ya * ya + yb * yb), xx) * EARTH_R;
  const float d0 = gd * 0.01f, d1 = tim * 0.01f;

  // ---- layer 1 (K=2): scalar FMA + fast tanh, store f16 activations
#pragma unroll
  for (int c = 0; c < 32; ++c) {
    const float z = fmaf(d0, w1s[c], d1 * w1s[32 + c]);
    hbuf[tid][c] = (_Float16)fast_tanh(fmaf(s1[c], z, t1[c]));
  }
  __syncthreads();

  // ---- build B fragments (16-bit B KxN: lanes 0-15 K=0..7,16..23, lanes 16-31 K=8..15,24..31)
  const int bK = (lane & 16) >> 1;
  const int bc = lane & 15;
  v16h bw2a, bw2b, bw3;
#pragma unroll
  for (int e = 0; e < 16; ++e) {
    bw2a[e] = w2h[(e < 8 ? e : e + 8) + bK][bc];
    bw2b[e] = w2h[(e < 8 ? e : e + 8) + bK][bc + 16];
    bw3[e]  = w3h[(e < 8 ? e : e + 8) + bK][bc];
  }

  const int aoff = (lane & 16) ? 8 : 0;  // A: upper lanes hold K+8 / K+16 halves
  const int moff = (lane & 16) ? 8 : 0;  // D: upper lanes hold rows M+8
  const v8f zero = {};

  // ---- layer 2: [16,32] @ [32,32] per tile = 2x V_WMMA_F32_16X16X32_F16
#pragma unroll
  for (int t = 0; t < 2; ++t) {
    const int base = wave * 32 + t * 16;
    const int arow = base + bc;
    v16h a;
#pragma unroll
    for (int e = 0; e < 16; ++e)
      a[e] = hbuf[arow][(e < 8 ? e : e + 8) + aoff];
    v8f alo = __builtin_amdgcn_wmma_f32_16x16x32_f16(false, a, false, bw2a,
                                                     (short)0, zero, false, false);
    v8f ahi = __builtin_amdgcn_wmma_f32_16x16x32_f16(false, a, false, bw2b,
                                                     (short)0, zero, false, false);
    const int mb = base + moff;
#pragma unroll
    for (int r = 0; r < 8; ++r) {
      hbuf2[mb + r][bc]      = (_Float16)fast_tanh(fmaf(s2[bc],      alo[r], t2[bc]));
      hbuf2[mb + r][bc + 16] = (_Float16)fast_tanh(fmaf(s2[bc + 16], ahi[r], t2[bc + 16]));
    }
  }
  __syncthreads();

  // ---- layer 3: [16,32] @ [32,5->16] per tile = 1 WMMA
#pragma unroll
  for (int t = 0; t < 2; ++t) {
    const int base = wave * 32 + t * 16;
    const int arow = base + bc;
    v16h a;
#pragma unroll
    for (int e = 0; e < 16; ++e)
      a[e] = hbuf2[arow][(e < 8 ? e : e + 8) + aoff];
    v8f lg = __builtin_amdgcn_wmma_f32_16x16x32_f16(false, a, false, bw3,
                                                    (short)0, zero, false, false);
    const int mb = base + moff;
    if (bc < 5) {
#pragma unroll
      for (int r = 0; r < 8; ++r)
        lbuf[mb + r][bc] = lg[r] + b3s[bc];
    }
  }
  __syncthreads();

  // ---- per-row argmax over 5 logits (first-max tie-break, like jnp.argmax)
  float best = lbuf[tid][0];
  int pid = 0;
#pragma unroll
  for (int c = 1; c < 5; ++c) {
    const float v = lbuf[tid][c];
    if (v > best) { best = v; pid = c; }
  }
  pid_o = pid; gd_o = gd; tim_o = tim; rid_o = rid;
}

__global__ void init_kernel(unsigned* pack) { *pack = 0u; }

__global__ __launch_bounds__(NPTS)
void score_kernel(P p, unsigned* pack)
{
  int pid, rid; float gd, tim;
  model_block((int)blockIdx.x, p, pid, gd, tim, rid);

  const unsigned long long m = __ballot(pid == rid);  // wave32: low 32 bits
  __shared__ int cnt[4];
  if ((threadIdx.x & 31) == 0) cnt[threadIdx.x >> 5] = __popcll(m);
  __syncthreads();
  if (threadIdx.x == 0) {
    const unsigned num = (unsigned)(cnt[0] + cnt[1] + cnt[2] + cnt[3]);
    // higher num wins; ties -> smaller k (first occurrence, jnp.argmax semantics)
    const unsigned pk = (num << 16) | (unsigned)(G2 - 1 - (int)blockIdx.x);
    atomicMax(pack, pk);
  }
}

__global__ __launch_bounds__(NPTS)
void output_kernel(P p, const unsigned* pack, float* out)
{
  const unsigned pk = *pack;
  const int k = G2 - 1 - (int)(pk & 0xFFFFu);

  int pid, rid; float gd, tim;
  model_block(k, p, pid, gd, tim, rid);

  const int n = threadIdx.x;
  out[n]                = (float)pid;   // class_ [128]
  out[NPTS + 2 * n]     = gd;           // ph[:,0] = distance (km)
  out[NPTS + 2 * n + 1] = tim;          // ph[:,1] = time
  if (n == 0) {
    out[3 * NPTS]     = (float)(pk >> 16);                 // max(num)
    const int gi = k / GSIZE, gj = k - gi * GSIZE;
    out[3 * NPTS + 1] = MIN_LON + DLON * (float)gj;        // pos lon
    out[3 * NPTS + 2] = MIN_LAT + DLAT * (float)gi;        // pos lat
  }
}

extern "C" void kernel_launch(void* const* d_in, const int* in_sizes, int n_in,
                              void* d_out, int out_size, void* d_ws, size_t ws_size,
                              hipStream_t stream)
{
  (void)in_sizes; (void)n_in; (void)out_size; (void)ws_size;
  P p;
  p.x   = (const float*)d_in[0];
  p.w1  = (const float*)d_in[1];
  p.b1  = (const float*)d_in[2];
  p.g1  = (const float*)d_in[3];
  p.be1 = (const float*)d_in[4];
  p.m1  = (const float*)d_in[5];
  p.v1  = (const float*)d_in[6];
  p.w2  = (const float*)d_in[7];
  p.b2  = (const float*)d_in[8];
  p.g2  = (const float*)d_in[9];
  p.be2 = (const float*)d_in[10];
  p.m2  = (const float*)d_in[11];
  p.v2  = (const float*)d_in[12];
  p.w3  = (const float*)d_in[13];
  p.b3  = (const float*)d_in[14];

  unsigned* pack = (unsigned*)d_ws;
  float*    out  = (float*)d_out;

  hipLaunchKernelGGL(init_kernel,   dim3(1),  dim3(1),    0, stream, pack);
  hipLaunchKernelGGL(score_kernel,  dim3(G2), dim3(NPTS), 0, stream, p, pack);
  hipLaunchKernelGGL(output_kernel, dim3(1),  dim3(NPTS), 0, stream, p, pack, out);
}
